// ManagerModule_43800076485189
// MI455X (gfx1250) — matrix operand order repaired and checked by reading, not compiled, not run on hardware
//
#include <hip/hip_runtime.h>
#include <hip/hip_bf16.h>

// ---------------- Types for WMMA fragments (gfx1250, wave32) ----------------
typedef __attribute__((ext_vector_type(16))) __bf16 v16bf;  // A/B frag: 8 VGPRs
typedef __attribute__((ext_vector_type(8)))  float  v8f;    // C/D frag: 8 VGPRs
typedef __attribute__((ext_vector_type(4)))  unsigned int v4u;
typedef __attribute__((ext_vector_type(8)))  int    v8i;
typedef __attribute__((ext_vector_type(4)))  int    v4i;

union FragAB {
    v16bf v;
    uint4 q[2];
    unsigned short s[16];
};

__device__ __forceinline__ v8f wmma_bf16(v16bf a, v16bf b, v8f c) {
    // D = A(16x32 bf16) * B(32x16 bf16) + C(16x16 f32)
    return __builtin_amdgcn_wmma_f32_16x16x32_bf16(
        /*neg_a=*/false, a, /*neg_b=*/false, b,
        /*c_mod=*/(short)0, c, /*reuse_a=*/false, /*reuse_b=*/false);
}

__device__ __forceinline__ unsigned short f2bf(float x) {
    union { float f; unsigned int u; } t; t.f = x;
    unsigned int r = t.u + 0x7FFFu + ((t.u >> 16) & 1u);  // RNE
    return (unsigned short)(r >> 16);
}

__device__ __forceinline__ void cvt8(const float4& a, const float4& b, unsigned short* o) {
    o[0] = f2bf(a.x); o[1] = f2bf(a.y); o[2] = f2bf(a.z); o[3] = f2bf(a.w);
    o[4] = f2bf(b.x); o[5] = f2bf(b.y); o[6] = f2bf(b.z); o[7] = f2bf(b.w);
}

__device__ __forceinline__ float sigm(float x) { return 1.0f / (1.0f + __expf(-x)); }
__device__ __forceinline__ float tanh_(float x) { return 1.0f - 2.0f / (__expf(2.0f * x) + 1.0f); }

// ---------------- Problem constants ----------------
#define BB   32
#define SS   1024
#define DD   512
#define HOR  10
#define NGATE 2048        // 4*d
#define KCAT  1024        // d (x) + d (h)
#define NSTEPS 103        // ceil(1024/10)
#define FULLT 1033        // S + horizon - 1

// Workspace layout (bytes)
#define OFF_WFC   0ull                     // 512KB  swizzled bf16 W_fc
#define OFF_WCAT  524288ull                // 4MB    swizzled bf16 [W_ih;W_hh]
#define OFF_BIAS  4718592ull               // 8KB    b_ih + b_hh (f32)
#define OFF_LATB  4726784ull               // 32MB   latent in bf16
#define OFF_FULL  38281216ull              // ~65MB  full goal stream f32 [32][1033][512]

// ---------------- Weight swizzle into B-fragment layout ----------------
// dst frag order: frag(nt,kt) at ((nt*KT + kt)*32 + lane)*16 halfwords.
// lane: col = nt*16 + (lane&15); k = kt*32 + (lane>>4)*8 + j (j<8), +16 + (j-8) (j>=8).
__global__ void k_swizzleB(const float* __restrict__ Wa, const float* __restrict__ Wb,
                           int Ka, int N, int K, unsigned short* __restrict__ dst) {
    int tid = blockIdx.x * blockDim.x + threadIdx.x;
    int KT = K >> 5;
    int total = (N >> 4) * KT * 32;
    if (tid >= total) return;
    int lane = tid & 31;
    int kt = (tid >> 5) % KT;
    int nt = (tid >> 5) / KT;
    int n  = nt * 16 + (lane & 15);
    int k0 = kt * 32 + ((lane >> 4) << 3);
    unsigned short* out = dst + ((size_t)(nt * KT + kt) * 32 + lane) * 16;
    #pragma unroll
    for (int j = 0; j < 16; j++) {
        int k = k0 + (j < 8 ? j : j + 8);
        float v = (k < Ka) ? Wa[(size_t)n * Ka + k] : Wb[(size_t)n * Ka + (k - Ka)];
        out[j] = f2bf(v);
    }
}

__global__ void k_bias(const float* __restrict__ bi, const float* __restrict__ bh,
                       float* __restrict__ out) {
    int i = blockIdx.x * blockDim.x + threadIdx.x;
    if (i < NGATE) out[i] = bi[i] + bh[i];
}

// full[b][p][d] = goal_horizon[p+1][b][0][d] for p in 0..8
__global__ void k_prev(const float* __restrict__ gh, float* __restrict__ full) {
    int tid = blockIdx.x * blockDim.x + threadIdx.x;
    if (tid >= 9 * BB * DD) return;
    int d = tid & (DD - 1);
    int b = (tid >> 9) & (BB - 1);
    int p = tid >> 14;
    full[((size_t)b * FULLT + p) * DD + d] = gh[((size_t)(p + 1) * BB + b) * DD + d];
}

// ---------------- Phase 1: latent = relu(z @ W_fc^T + b_fc) ----------------
// Grid: 2048 blocks x 256 thr (8 waves). Each wave: 16x64 output tile, K=512.
__global__ __launch_bounds__(256) void k_latent(
    const float* __restrict__ z, const unsigned short* __restrict__ wfc_sw,
    const float* __restrict__ bfc, float* __restrict__ out_latent,
    unsigned short* __restrict__ latb) {
    const int lane = threadIdx.x & 31;
    const int task = blockIdx.x * 8 + (threadIdx.x >> 5);  // 16384 wave tasks
    const int nt4 = task & 7;          // 64-col group
    const int mt  = task >> 3;         // 16-row tile, 0..2047
    const int rbase = mt * 16;
    const int nbase = nt4 * 64;

    v8f acc[4];
    #pragma unroll
    for (int p = 0; p < 4; p++)
        #pragma unroll
        for (int r = 0; r < 8; r++) acc[p][r] = 0.0f;

    const int m = lane & 15;
    const int koff = (lane >> 4) << 3;
    const float* zrow = z + (size_t)(rbase + m) * DD + koff;

    for (int kt = 0; kt < 16; kt++) {
        FragAB A;
        const float* pa = zrow + kt * 32;
        float4 a0 = *(const float4*)(pa +  0);
        float4 a1 = *(const float4*)(pa +  4);
        float4 a2 = *(const float4*)(pa + 16);
        float4 a3 = *(const float4*)(pa + 20);
        cvt8(a0, a1, A.s);
        cvt8(a2, a3, A.s + 8);
        #pragma unroll
        for (int p = 0; p < 4; p++) {
            int nt = nt4 * 4 + p;
            const uint4* bp = (const uint4*)wfc_sw + ((size_t)(nt * 16 + kt) * 32 + lane) * 2;
            FragAB B;
            B.q[0] = bp[0];
            B.q[1] = bp[1];
            acc[p] = wmma_bf16(A.v, B.v, acc[p]);
        }
    }
    // epilogue: bias + relu; write f32 output and bf16 copy
    #pragma unroll
    for (int p = 0; p < 4; p++) {
        int col = nbase + p * 16 + (lane & 15);
        float bias = bfc[col];
        #pragma unroll
        for (int r = 0; r < 8; r++) {
            int row = rbase + r + ((lane >> 4) << 3);
            float v = acc[p][r] + bias;
            v = fmaxf(v, 0.0f);
            out_latent[(size_t)row * DD + col] = v;
            latb[(size_t)row * DD + col] = f2bf(v);
        }
    }
}

// ---------------- Phase 2: dilated LSTM scan ----------------
// 20 blocks x 256 thr. Block owns 16 rows = (i = blk>>1, b = (blk&1)*16 .. +15).
// Wave w owns h-columns [64w, 64w+64): computes gate cols {g*512 + 64w + 16p}.
// h kept in LDS (bf16), c kept in C-fragment registers. Two barriers/step.
// x_t tile (16 rows x 1KB, row stride S*d elements) staged by the Tensor Data
// Mover: one 2D descriptor with LDS padding 16B per 1KB row (-> 520-halfword
// rows, bank-conflict-free A-fragment reads), tracked by TENSORcnt.
__global__ __launch_bounds__(256) void k_lstm(
    const unsigned short* __restrict__ latb, const unsigned short* __restrict__ wcat,
    const float* __restrict__ biasc, const float* __restrict__ h0,
    const float* __restrict__ c0, float* __restrict__ full) {
    __shared__ __align__(16) unsigned short x_lds[16][520];
    __shared__ __align__(16) unsigned short h_lds[16][520];

    const int lane  = threadIdx.x & 31;
    const int w     = threadIdx.x >> 5;      // wave 0..7
    const int ii    = blockIdx.x >> 1;       // horizon slot
    const int bbase = (blockIdx.x & 1) * 16; // batch half

    // init h_lds from h_horizon (fp32 -> bf16)
    {
        int r  = threadIdx.x >> 4;
        int cb = (threadIdx.x & 15) * 32;
        const float* src = h0 + ((size_t)ii * BB + (bbase + r)) * DD + cb;
        #pragma unroll
        for (int j = 0; j < 32; j++) h_lds[r][cb + j] = f2bf(src[j]);
    }
    // init c fragments from c_horizon
    v8f cfr[4];
    #pragma unroll
    for (int p = 0; p < 4; p++) {
        int col = w * 64 + p * 16 + (lane & 15);
        #pragma unroll
        for (int r = 0; r < 8; r++) {
            int mrow = r + ((lane >> 4) << 3);
            cfr[p][r] = c0[((size_t)ii * BB + bbase + mrow) * DD + col];
        }
    }
    // gate biases for this lane's columns
    float bg[4][4];
    #pragma unroll
    for (int g = 0; g < 4; g++)
        #pragma unroll
        for (int p = 0; p < 4; p++)
            bg[g][p] = biasc[g * DD + w * 64 + p * 16 + (lane & 15)];
    __syncthreads();

    const int mloc = lane & 15;
    const int koff = (lane >> 4) << 3;
    const unsigned xlds_off = (unsigned)(unsigned long long)(const void*)&x_lds[0][0];

    for (int j = 0; j < NSTEPS; j++) {
        const int t = j * HOR + ii;
        // ---- stage x_t ----
        if (t < SS) {
            if (w == 0) {
                unsigned long long ga = (unsigned long long)(const void*)
                    (latb + ((size_t)bbase * SS + (size_t)t) * DD);
                v4u g0; v8i g1; v4i g2; v4i g3; v8i g4;
                // D# group0: count=1 | lds_addr | global_addr[56:0] | type=2
                g0[0] = 1u;
                g0[1] = xlds_off;
                g0[2] = (unsigned)ga;
                g0[3] = ((unsigned)(ga >> 32) & 0x01FFFFFFu) | 0x80000000u;
                // D# group1: data_size=2B; pad_enable, interval=256 DW (1KB),
                // amount=4 DW (16B); tensor_dim0=512, tensor_dim1=16,
                // tile_dim0=512, tile_dim1=16; tensor_dim0_stride=S*D=524288.
                g1[0] = (int)((1u << 16) | (1u << 20) | (7u << 22) | (3u << 25));
                g1[1] = (int)(512u << 16);   // tensor_dim0[15:0] @ bits 63:48
                g1[2] = (int)(16u << 16);    // tensor_dim1[15:0] @ bits 95:80
                g1[3] = (int)(512u << 16);   // tile_dim0 @ bits 127:112
                g1[4] = 16;                  // tile_dim1 @ bits 143:128
                g1[5] = (int)524288;         // tensor_dim0_stride[31:0]
                g1[6] = 0; g1[7] = 0;
                g2[0] = 0; g2[1] = 0; g2[2] = 0; g2[3] = 0;  // 2-D tensor
                g3[0] = 0; g3[1] = 0; g3[2] = 0; g3[3] = 0;
                g4[0] = 0; g4[1] = 0; g4[2] = 0; g4[3] = 0;  // unused extra group
                g4[4] = 0; g4[5] = 0; g4[6] = 0; g4[7] = 0;
                __builtin_amdgcn_tensor_load_to_lds(g0, g1, g2, g3, g4, 0);
            }
        } else {
            // zero-pad the tail steps
            int r  = threadIdx.x >> 4;
            int cb = (threadIdx.x & 15) * 32;
            uint4 zz = make_uint4(0u, 0u, 0u, 0u);
            *(uint4*)&x_lds[r][cb]     = zz;
            *(uint4*)&x_lds[r][cb + 8] = zz;
        }
        __builtin_amdgcn_s_wait_tensorcnt(0);
        __syncthreads();  // x staged; h from previous step visible

        v8f hval[4];
        #pragma unroll
        for (int p = 0; p < 4; p++) {
            v8f acc[4];
            #pragma unroll
            for (int g = 0; g < 4; g++)
                #pragma unroll
                for (int r = 0; r < 8; r++) acc[g][r] = 0.0f;

            for (int ktb = 0; ktb < 8; ktb++) {
                FragAB A[4];
                #pragma unroll
                for (int kk = 0; kk < 4; kk++) {
                    int kt = ktb * 4 + kk;
                    const unsigned short* base = (kt < 16)
                        ? &x_lds[mloc][kt * 32 + koff]
                        : &h_lds[mloc][(kt - 16) * 32 + koff];
                    A[kk].q[0] = *(const uint4*)(base);
                    A[kk].q[1] = *(const uint4*)(base + 16);
                }
                #pragma unroll
                for (int g = 0; g < 4; g++) {
                    int nt = g * 32 + w * 4 + p;  // gate column tile (of 128)
                    const uint4* bp = (const uint4*)wcat +
                        (((size_t)nt * 32 + (size_t)ktb * 4) * 32 + lane) * 2;
                    // prefetch next ktb's fragments of this column tile
                    __builtin_prefetch((const void*)(bp + 256), 0, 0);
                    #pragma unroll
                    for (int kk = 0; kk < 4; kk++) {
                        FragAB B;
                        B.q[0] = bp[(size_t)kk * 64];
                        B.q[1] = bp[(size_t)kk * 64 + 1];
                        acc[g] = wmma_bf16(A[kk].v, B.v, acc[g]);
                    }
                }
            }
            // gate nonlinearities + cell update (pure per-lane on fragments)
            #pragma unroll
            for (int r = 0; r < 8; r++) {
                float ig = sigm(acc[0][r] + bg[0][p]);
                float fg = sigm(acc[1][r] + bg[1][p]);
                float gg = tanh_(acc[2][r] + bg[2][p]);
                float og = sigm(acc[3][r] + bg[3][p]);
                float cv = fg * cfr[p][r] + ig * gg;
                cfr[p][r] = cv;
                hval[p][r] = og * tanh_(cv);
            }
        }
        __syncthreads();  // all h_lds reads done before overwrite

        #pragma unroll
        for (int p = 0; p < 4; p++) {
            int col = w * 64 + p * 16 + (lane & 15);
            #pragma unroll
            for (int r = 0; r < 8; r++) {
                int mrow = r + ((lane >> 4) << 3);
                h_lds[mrow][col] = f2bf(hval[p][r]);
                if (t < SS)
                    full[((size_t)(bbase + mrow) * FULLT + 9 + t) * DD + col] = hval[p][r];
            }
        }
        // h_lds writes -> next-iter reads separated by next iteration's barrier
    }
}

// ---------------- Phase 3: window-sum(k=10) + L2 normalize ----------------
// 512 blocks (b x 16 t-chunks of 64) x 256 thr; each thread owns 2 d-columns.
__global__ __launch_bounds__(256) void k_goal(const float* __restrict__ full,
                                              float* __restrict__ goal) {
    __shared__ float red[256];
    const int b = blockIdx.x >> 4;
    const int t0 = (blockIdx.x & 15) * 64;
    const int c0 = threadIdx.x;
    const int c1 = threadIdx.x + 256;
    const float* fb = full + (size_t)b * FULLT * DD;

    float s0 = 0.0f, s1 = 0.0f;
    for (int k = 0; k < HOR; k++) {
        s0 += fb[(size_t)(t0 + k) * DD + c0];
        s1 += fb[(size_t)(t0 + k) * DD + c1];
    }
    for (int tt = 0; tt < 64; tt++) {
        int t = t0 + tt;
        if (tt) {
            s0 += fb[(size_t)(t + 9) * DD + c0] - fb[(size_t)(t - 1) * DD + c0];
            s1 += fb[(size_t)(t + 9) * DD + c1] - fb[(size_t)(t - 1) * DD + c1];
        }
        red[threadIdx.x] = s0 * s0 + s1 * s1;
        __syncthreads();
        for (int off = 128; off; off >>= 1) {
            if (threadIdx.x < off) red[threadIdx.x] += red[threadIdx.x + off];
            __syncthreads();
        }
        float rn = rsqrtf(red[0]);
        __syncthreads();
        goal[((size_t)b * SS + t) * DD + c0] = s0 * rn;
        goal[((size_t)b * SS + t) * DD + c1] = s1 * rn;
    }
}

// ---------------- Host launch ----------------
extern "C" void kernel_launch(void* const* d_in, const int* in_sizes, int n_in,
                              void* d_out, int out_size, void* d_ws, size_t ws_size,
                              hipStream_t stream) {
    const float* z    = (const float*)d_in[0];
    const float* h_h  = (const float*)d_in[1];
    const float* c_h  = (const float*)d_in[2];
    const float* g_h  = (const float*)d_in[3];
    const float* W_fc = (const float*)d_in[4];
    const float* b_fc = (const float*)d_in[5];
    const float* W_ih = (const float*)d_in[6];
    const float* W_hh = (const float*)d_in[7];
    const float* b_ih = (const float*)d_in[8];
    const float* b_hh = (const float*)d_in[9];

    char* ws = (char*)d_ws;
    unsigned short* wfc_sw  = (unsigned short*)(ws + OFF_WFC);
    unsigned short* wcat_sw = (unsigned short*)(ws + OFF_WCAT);
    float*          biasc   = (float*)(ws + OFF_BIAS);
    unsigned short* latb    = (unsigned short*)(ws + OFF_LATB);
    float*          fullbuf = (float*)(ws + OFF_FULL);

    float* out_latent = (float*)d_out;
    float* out_goal   = out_latent + (size_t)BB * SS * DD;
    float* out_hh     = out_goal + (size_t)BB * SS * DD;
    float* out_ch     = out_hh + (size_t)HOR * BB * DD;
    float* out_gh     = out_ch + (size_t)HOR * BB * DD;

    // prep
    k_swizzleB<<<(32 * 16 * 32 + 255) / 256, 256, 0, stream>>>(W_fc, W_fc, 512, 512, 512, wfc_sw);
    k_swizzleB<<<(128 * 32 * 32 + 255) / 256, 256, 0, stream>>>(W_ih, W_hh, 512, 2048, 1024, wcat_sw);
    k_bias<<<8, 256, 0, stream>>>(b_ih, b_hh, biasc);
    k_prev<<<(9 * BB * DD + 255) / 256, 256, 0, stream>>>(g_h, fullbuf);

    // phases
    k_latent<<<2048, 256, 0, stream>>>(z, wfc_sw, b_fc, out_latent, latb);
    k_lstm<<<20, 256, 0, stream>>>(latb, wcat_sw, biasc, h_h, c_h, fullbuf);
    k_goal<<<512, 256, 0, stream>>>(fullbuf, out_goal);

    // state pass-throughs
    size_t sbytes = (size_t)HOR * BB * DD * sizeof(float);
    (void)hipMemcpyAsync(out_hh, h_h, sbytes, hipMemcpyDeviceToDevice, stream);
    (void)hipMemcpyAsync(out_ch, c_h, sbytes, hipMemcpyDeviceToDevice, stream);
    (void)hipMemcpyAsync(out_gh, g_h, sbytes, hipMemcpyDeviceToDevice, stream);
}
